// DogDetector_77129022701615
// MI455X (gfx1250) — matrix operand order repaired and checked by reading, not compiled, not run on hardware
//
#include <hip/hip_runtime.h>
#include <hip/hip_bf16.h>
#include <math.h>

typedef __bf16 bf16;
typedef __attribute__((ext_vector_type(16))) __bf16 v16bf;
typedef __attribute__((ext_vector_type(8)))  __bf16 v8bf;
typedef __attribute__((ext_vector_type(8)))  float   v8f;
typedef __attribute__((ext_vector_type(8)))  unsigned short u16x8;

__device__ __forceinline__ bf16 f2bf(float f) {
    union { float f; unsigned u; } v; v.f = f;
    unsigned r = v.u + 0x7FFFu + ((v.u >> 16) & 1u);   // round-to-nearest-even
    unsigned short h = (unsigned short)(r >> 16);
    return __builtin_bit_cast(__bf16, h);
}

// ---------------------------------------------------------------------------
// Weight / bias preparation: fold BN scale into bf16-packed GEMM weights.
// Layout: Wp[oc][K] with K = ic*KS*KS + ky*KS + kx  (== flat OIHW order).
// ---------------------------------------------------------------------------
__global__ void prep_w_bn(const float* __restrict__ w, const float* __restrict__ g,
                          const float* __restrict__ var, bf16* __restrict__ Wp,
                          int K, int total) {
    int i = blockIdx.x * 256 + threadIdx.x;
    if (i >= total) return;
    int oc = i / K;
    float inv = g[oc] * rsqrtf(var[oc] + 1e-5f);
    Wp[i] = f2bf(w[i] * inv);
}

__global__ void prep_bias_bn(const float* __restrict__ b, const float* __restrict__ g,
                             const float* __restrict__ be, const float* __restrict__ mu,
                             const float* __restrict__ var, float* __restrict__ out, int M) {
    int i = blockIdx.x * 256 + threadIdx.x;
    if (i < M) {
        float inv = g[i] * rsqrtf(var[i] + 1e-5f);
        out[i] = b[i] * inv + be[i] - mu[i] * inv;
    }
}

// Head: rows 0..47 = box weights, 48..59 = cls weights, 60..63 = zero pad.
__global__ void prep_head_w(const float* __restrict__ wbox, const float* __restrict__ wcls,
                            bf16* __restrict__ Wp) {
    int i = blockIdx.x * 256 + threadIdx.x;
    if (i >= 64 * 2304) return;
    int oc = i / 2304, k = i - oc * 2304;
    float v = 0.0f;
    if (oc < 48)      v = wbox[oc * 2304 + k];
    else if (oc < 60) v = wcls[(oc - 48) * 2304 + k];
    Wp[i] = f2bf(v);
}

__global__ void prep_head_bias(const float* __restrict__ bbox, const float* __restrict__ bcls,
                               float* __restrict__ out) {
    int i = threadIdx.x;
    if (i < 64) {
        float v = 0.0f;
        if (i < 48)      v = bbox[i];
        else if (i < 60) v = bcls[i - 48];
        out[i] = v;
    }
}

// ---------------------------------------------------------------------------
// Conv (1x1 or 3x3, pad=KS/2) as implicit GEMM with bf16 WMMA.
// One WG (8 waves) per image.
//   MTOT=256: wave w -> m-tiles {2w,2w+1} x n-tiles {0..3}  (8 acc tiles/wave)
//   MTOT=64 : wave w -> m-tile  {w>>1}   x n-tiles {(w&1)*2,+1}
// K processed in chunks of 128 (4 WMMA K-steps per chunk).
// Activations (KS==3 inputs, !HEAD outputs) use [n][49][C] layout.
// conv1 (KS==1) reads features [n][512][49] directly from global.
// ---------------------------------------------------------------------------
template<int CIN, int KS, int MTOT, int HEAD>
__global__ __launch_bounds__(256)
void conv_bn_wmma(const float* __restrict__ X,
                  const bf16* __restrict__ Wp,    // [MTOT][K] bf16
                  const float* __restrict__ bias, // [MTOT]
                  float* __restrict__ Y,          // !HEAD: [B][49][MTOT]
                  float* __restrict__ Ybox,       // HEAD: [B][49][48]
                  float* __restrict__ Yconf) {    // HEAD: [B][49][12]
    constexpr int K   = CIN * KS * KS;
    constexpr int KC  = 128;
    constexpr int NKC = K / KC;
    constexpr int BTS = 136;                       // Bt row stride (bank-safe, 16B-aligned)
    constexpr int NMT = (MTOT == 256) ? 2 : 1;
    constexpr int NNT = (MTOT == 256) ? 4 : 2;

    __shared__ __align__(16) bf16 sBt[64 * BTS];
    __shared__ __align__(16) bf16 sX[(KS == 3) ? CIN * 81 : 8];
    __shared__ __align__(16) unsigned short sOffAll[K];

    const int n    = blockIdx.x;
    const int tid  = threadIdx.x;
    const int lane = tid & 31;
    const int wv   = tid >> 5;
    const int half = lane >> 4;
    const int l16  = lane & 15;
    const int mtb  = (MTOT == 256) ? wv * 2 : (wv >> 1);
    const int ntb  = (MTOT == 256) ? 0 : (wv & 1) * 2;

    const float* Xn = X + (size_t)n * CIN * 49;

    // ---- one-time: im2col offset table (u16) + zero sX halo ----
    for (int i = tid; i < K; i += 256) {
        int off;
        if (KS == 3) {
            int ic = i / 9, r = i - ic * 9;
            int ry = r / 3, rx = r - ry * 3;
            off = ic * 81 + ry * 9 + rx;
        } else {
            off = i * 49;
        }
        sOffAll[i] = (unsigned short)off;
    }
    if (KS == 3) {
        bf16 z = f2bf(0.0f);
        v8bf zv = {z, z, z, z, z, z, z, z};
        for (int i = tid; i < CIN * 81 / 8; i += 256)
            *(v8bf*)&sX[i * 8] = zv;
        __syncthreads();
        // coalesced stage of [49][CIN] activations into padded 9x9 LDS image
        for (int idx = tid; idx < CIN * 49; idx += 256) {
            int pix = idx / CIN, ic = idx - pix * CIN;
            int y = pix / 7, x = pix - y * 7;
            sX[ic * 81 + (y + 1) * 9 + (x + 1)] = f2bf(Xn[idx]);
        }
    }
    __syncthreads();

    v8f acc[NMT][NNT] = {};
    const bf16* wr[NMT];
    #pragma unroll
    for (int im = 0; im < NMT; ++im)
        wr[im] = Wp + (size_t)((mtb + im) * 16 + l16) * K;

    for (int kc = 0; kc < NKC; ++kc) {
        // ---- build Bt[64 cols][128 kk] via offset table (b128 stores) ----
        #pragma unroll
        for (int it = 0; it < 4; ++it) {
            int g   = tid + it * 256;          // 0..1023
            int p   = g >> 4;                  // column 0..63
            int kk0 = (g & 15) << 3;           // 0,8,...,120
            bf16 z = f2bf(0.0f);
            v8bf v = {z, z, z, z, z, z, z, z};
            if (p < 49) {
                u16x8 offs = *(const u16x8*)&sOffAll[kc * KC + kk0];
                int py = p / 7;
                int pq = (KS == 3) ? (py * 9 + (p - py * 7)) : p;
                #pragma unroll
                for (int e = 0; e < 8; ++e) {
                    int o = (int)offs[e] + pq;
                    v[e] = (KS == 3) ? sX[o] : f2bf(Xn[o]);
                }
            }
            *(v8bf*)&sBt[p * BTS + kk0] = v;
        }
        __syncthreads();

        // ---- 4 WMMA K-steps over this chunk ----
        #pragma unroll
        for (int sub = 0; sub < 4; ++sub) {
            // A fragments (ISA layout: e -> K = (e>=8?16:0) + half*8 + (e&7))
            v16bf a[NMT];
            #pragma unroll
            for (int im = 0; im < NMT; ++im) {
                const bf16* ab = wr[im] + kc * KC + sub * 32 + half * 8;
                v8bf a0 = *(const v8bf*)(ab);
                v8bf a1 = *(const v8bf*)(ab + 16);
                #pragma unroll
                for (int e = 0; e < 8; ++e) { a[im][e] = a0[e]; a[im][e + 8] = a1[e]; }
            }
            #pragma unroll
            for (int in_ = 0; in_ < NNT; ++in_) {
                int nt = ntb + in_;
                const bf16* bp = &sBt[(nt * 16 + l16) * BTS + sub * 32 + half * 8];
                v8bf bl = *(const v8bf*)(bp);
                v8bf bh = *(const v8bf*)(bp + 16);
                v16bf b;
                #pragma unroll
                for (int e = 0; e < 8; ++e) { b[e] = bl[e]; b[e + 8] = bh[e]; }
                #pragma unroll
                for (int im = 0; im < NMT; ++im)
                    acc[im][in_] = __builtin_amdgcn_wmma_f32_16x16x32_bf16(
                        false, a[im], false, b, (short)0, acc[im][in_], false, false);
            }
        }
        __syncthreads();
    }

    // ---- epilogue: C layout M = r + half*8 (contiguous channels), N = l16 ----
    #pragma unroll
    for (int im = 0; im < NMT; ++im) {
        int m0 = (mtb + im) * 16 + half * 8;
        float bs[8];
        #pragma unroll
        for (int r = 0; r < 8; ++r) bs[r] = bias[m0 + r];
        #pragma unroll
        for (int in_ = 0; in_ < NNT; ++in_) {
            int N = (ntb + in_) * 16 + l16;
            if (N < 49) {
                if (HEAD == 0) {
                    float4 lo, hi;
                    lo.x = fmaxf(acc[im][in_][0] + bs[0], 0.0f);
                    lo.y = fmaxf(acc[im][in_][1] + bs[1], 0.0f);
                    lo.z = fmaxf(acc[im][in_][2] + bs[2], 0.0f);
                    lo.w = fmaxf(acc[im][in_][3] + bs[3], 0.0f);
                    hi.x = fmaxf(acc[im][in_][4] + bs[4], 0.0f);
                    hi.y = fmaxf(acc[im][in_][5] + bs[5], 0.0f);
                    hi.z = fmaxf(acc[im][in_][6] + bs[6], 0.0f);
                    hi.w = fmaxf(acc[im][in_][7] + bs[7], 0.0f);
                    float* yp = Y + ((size_t)n * 49 + N) * MTOT + m0;
                    *(float4*)(yp)     = lo;
                    *(float4*)(yp + 4) = hi;
                } else {
                    #pragma unroll
                    for (int r = 0; r < 8; ++r) {
                        int m = m0 + r;
                        float v = acc[im][in_][r] + bs[r];
                        if (m < 48)
                            Ybox[((size_t)n * 49 + N) * 48 + m] = v;
                        else if (m < 60)
                            Yconf[((size_t)n * 49 + N) * 12 + (m - 48)] =
                                1.0f / (1.0f + __expf(-v));
                    }
                }
            }
        }
    }
}

// ---------------------------------------------------------------------------
// Per-image anchor decode + bitonic sort + greedy NMS + top-100 compaction.
// ---------------------------------------------------------------------------
#define NANCH 588
#define SORTN 1024

__global__ __launch_bounds__(256)
void decode_nms_kernel(const float* __restrict__ boxp,   // [B][49][48]
                       const float* __restrict__ confp,  // [B][49][12]
                       float* __restrict__ out) {        // [B][100][6]
    __shared__ float sScore[SORTN];
    __shared__ int   sIdx[SORTN];
    __shared__ float sBox[NANCH * 4];
    __shared__ unsigned char sSup[NANCH];
    __shared__ unsigned char sKeep[NANCH];
    __shared__ float sCur[4];
    __shared__ int   sCurKeep;

    const int n   = blockIdx.x;
    const int tid = threadIdx.x;
    const float SC[4] = {0.3f, 0.5f, 0.7f, 0.9f};
    const float RT[3] = {0.7f, 1.0f, 1.3f};

    for (int a = tid; a < SORTN; a += 256) {
        if (a < NANCH) {
            int p = a / 12, d = a - p * 12;
            float cx = ((float)(p / 7) + 0.5f) * (1.0f / 7.0f);
            float cy = ((float)(p % 7) + 0.5f) * (1.0f / 7.0f);
            float sc = SC[d / 3];
            float rt = sqrtf(RT[d % 3]);
            float ws = sc * rt, hs = sc / rt;
            const float* bp = boxp + ((size_t)n * 49 + p) * 48 + d * 4;
            float pcx = bp[0] * ws + cx;
            float pcy = bp[1] * hs + cy;
            float psw = __expf(bp[2]) * ws;
            float psh = __expf(bp[3]) * hs;
            sBox[a * 4 + 0] = pcx - 0.5f * psw;
            sBox[a * 4 + 1] = pcy - 0.5f * psh;
            sBox[a * 4 + 2] = pcx + 0.5f * psw;
            sBox[a * 4 + 3] = pcy + 0.5f * psh;
            sScore[a] = confp[((size_t)n * 49 + p) * 12 + d];
            sSup[a] = 0; sKeep[a] = 0;
            sIdx[a] = a;
        } else {
            sScore[a] = -1.0f; sIdx[a] = a;
        }
    }
    __syncthreads();

    // bitonic sort, descending by score
    for (int k = 2; k <= SORTN; k <<= 1) {
        for (int j = k >> 1; j > 0; j >>= 1) {
            for (int i = tid; i < SORTN; i += 256) {
                int p = i ^ j;
                if (p > i) {
                    bool desc = ((i & k) == 0);
                    float si = sScore[i], sp = sScore[p];
                    bool sw = desc ? (si < sp) : (si > sp);
                    if (sw) {
                        sScore[i] = sp; sScore[p] = si;
                        int t = sIdx[i]; sIdx[i] = sIdx[p]; sIdx[p] = t;
                    }
                }
            }
            __syncthreads();
        }
    }

    // greedy NMS over sorted order (suppress j > i with IoU > 0.5)
    for (int i = 0; i < NANCH; ++i) {
        if (tid == 0) {
            int keep = (sScore[i] > 0.3f) && (!sSup[i]);
            sKeep[i] = (unsigned char)keep;
            sCurKeep = keep;
            if (keep) {
                int bi = sIdx[i];
                sCur[0] = sBox[bi * 4 + 0]; sCur[1] = sBox[bi * 4 + 1];
                sCur[2] = sBox[bi * 4 + 2]; sCur[3] = sBox[bi * 4 + 3];
            }
        }
        __syncthreads();
        if (sCurKeep) {
            float x0 = sCur[0], y0 = sCur[1], x1 = sCur[2], y1 = sCur[3];
            float areaI = (x1 - x0) * (y1 - y0);
            for (int j = i + 1 + tid; j < NANCH; j += 256) {
                if (!sSup[j]) {
                    int bj = sIdx[j];
                    float bx0 = sBox[bj * 4], by0 = sBox[bj * 4 + 1];
                    float bx1 = sBox[bj * 4 + 2], by1 = sBox[bj * 4 + 3];
                    float areaJ = (bx1 - bx0) * (by1 - by0);
                    float lx = fmaxf(x0, bx0), ly = fmaxf(y0, by0);
                    float rx = fminf(x1, bx1), ry = fminf(y1, by1);
                    float w = fmaxf(rx - lx, 0.0f), h = fmaxf(ry - ly, 0.0f);
                    float inter = w * h;
                    float iou = inter / (areaI + areaJ - inter + 1e-9f);
                    if (iou > 0.5f) sSup[j] = 1;
                }
            }
        }
        __syncthreads();
    }

    // compact top-100 kept (already score-descending)
    for (int t = tid; t < 600; t += 256) out[(size_t)n * 600 + t] = 0.0f;
    __syncthreads();
    if (tid == 0) {
        int cnt = 0;
        for (int i = 0; i < NANCH && cnt < 100; ++i) {
            if (sKeep[i]) {
                int bi = sIdx[i];
                float* o = out + (size_t)n * 600 + cnt * 6;
                o[0] = sBox[bi * 4 + 0]; o[1] = sBox[bi * 4 + 1];
                o[2] = sBox[bi * 4 + 2]; o[3] = sBox[bi * 4 + 3];
                o[4] = sScore[i]; o[5] = 1.0f;
                ++cnt;
            }
        }
    }
}

// ---------------------------------------------------------------------------
extern "C" void kernel_launch(void* const* d_in, const int* in_sizes, int n_in,
                              void* d_out, int out_size, void* d_ws, size_t ws_size,
                              hipStream_t stream) {
    (void)in_sizes; (void)n_in; (void)out_size; (void)ws_size;
    const float* features = (const float*)d_in[0];
    const float *w1 = (const float*)d_in[1],  *b1 = (const float*)d_in[2],
                *g1 = (const float*)d_in[3],  *be1 = (const float*)d_in[4],
                *mu1 = (const float*)d_in[5], *var1 = (const float*)d_in[6];
    const float *w2 = (const float*)d_in[7],  *b2 = (const float*)d_in[8],
                *g2 = (const float*)d_in[9],  *be2 = (const float*)d_in[10],
                *mu2 = (const float*)d_in[11], *var2 = (const float*)d_in[12];
    const float *w3 = (const float*)d_in[13], *b3 = (const float*)d_in[14],
                *g3 = (const float*)d_in[15], *be3 = (const float*)d_in[16],
                *mu3 = (const float*)d_in[17], *var3 = (const float*)d_in[18];
    const float *w4 = (const float*)d_in[19], *b4 = (const float*)d_in[20],
                *g4 = (const float*)d_in[21], *be4 = (const float*)d_in[22],
                *mu4 = (const float*)d_in[23], *var4 = (const float*)d_in[24];
    const float *w_cls = (const float*)d_in[25], *b_cls = (const float*)d_in[26];
    const float *w_box = (const float*)d_in[27], *b_box = (const float*)d_in[28];

    char* ws = (char*)d_ws;
    size_t off = 0;
    auto take = [&](size_t bytes) -> void* {
        void* p = ws + off;
        off = (off + bytes + 255) & ~(size_t)255;
        return p;
    };
    constexpr int B = 256;
    bf16*  Wp1   = (bf16*)take((size_t)256 * 512 * 2);
    bf16*  Wp2   = (bf16*)take((size_t)256 * 2304 * 2);
    bf16*  Wp3   = (bf16*)take((size_t)256 * 2304 * 2);
    bf16*  Wp4   = (bf16*)take((size_t)256 * 2304 * 2);
    bf16*  Wph   = (bf16*)take((size_t)64 * 2304 * 2);
    float* bias1 = (float*)take(256 * 4);
    float* bias2 = (float*)take(256 * 4);
    float* bias3 = (float*)take(256 * 4);
    float* bias4 = (float*)take(256 * 4);
    float* biash = (float*)take(64 * 4);
    float* actA  = (float*)take((size_t)B * 49 * 256 * 4);
    float* actB  = (float*)take((size_t)B * 49 * 256 * 4);
    float* boxb  = (float*)take((size_t)B * 49 * 48 * 4);
    float* confb = (float*)take((size_t)B * 49 * 12 * 4);

    // weight / bias prep
    prep_w_bn<<<(256 * 512 + 255) / 256, 256, 0, stream>>>(w1, g1, var1, Wp1, 512, 256 * 512);
    prep_w_bn<<<(256 * 2304 + 255) / 256, 256, 0, stream>>>(w2, g2, var2, Wp2, 2304, 256 * 2304);
    prep_w_bn<<<(256 * 2304 + 255) / 256, 256, 0, stream>>>(w3, g3, var3, Wp3, 2304, 256 * 2304);
    prep_w_bn<<<(256 * 2304 + 255) / 256, 256, 0, stream>>>(w4, g4, var4, Wp4, 2304, 256 * 2304);
    prep_head_w<<<(64 * 2304 + 255) / 256, 256, 0, stream>>>(w_box, w_cls, Wph);
    prep_bias_bn<<<1, 256, 0, stream>>>(b1, g1, be1, mu1, var1, bias1, 256);
    prep_bias_bn<<<1, 256, 0, stream>>>(b2, g2, be2, mu2, var2, bias2, 256);
    prep_bias_bn<<<1, 256, 0, stream>>>(b3, g3, be3, mu3, var3, bias3, 256);
    prep_bias_bn<<<1, 256, 0, stream>>>(b4, g4, be4, mu4, var4, bias4, 256);
    prep_head_bias<<<1, 64, 0, stream>>>(b_box, b_cls, biash);

    // conv pipeline (WMMA implicit GEMM), one WG per image
    conv_bn_wmma<512, 1, 256, 0><<<B, 256, 0, stream>>>(features, Wp1, bias1, actA, nullptr, nullptr);
    conv_bn_wmma<256, 3, 256, 0><<<B, 256, 0, stream>>>(actA, Wp2, bias2, actB, nullptr, nullptr);
    conv_bn_wmma<256, 3, 256, 0><<<B, 256, 0, stream>>>(actB, Wp3, bias3, actA, nullptr, nullptr);
    conv_bn_wmma<256, 3, 256, 0><<<B, 256, 0, stream>>>(actA, Wp4, bias4, actB, nullptr, nullptr);
    conv_bn_wmma<256, 3, 64, 1><<<B, 256, 0, stream>>>(actB, Wph, biash, nullptr, boxb, confb);

    // decode + NMS
    decode_nms_kernel<<<B, 256, 0, stream>>>(boxb, confb, (float*)d_out);
}
